// TopKSAE_42219528520070
// MI455X (gfx1250) — matrix operand order repaired and checked by reading, not compiled, not run on hardware
//
#include <hip/hip_runtime.h>
#include <hip/hip_bf16.h>

// ---------------- problem constants ----------------
#define BATCH 32768
#define DIN   256      // input dim D
#define NF    4096     // n_features N
#define TOPK  32       // top-k
#define TILE_M 16      // batch rows per workgroup (one WMMA M tile)

typedef __attribute__((ext_vector_type(16))) __bf16 v16bf;
typedef __attribute__((ext_vector_type(8)))  __bf16 v8bf;
typedef __attribute__((ext_vector_type(8)))  float  v8f;

// ---------------- LDS layout (dynamic) ----------------
#define A_STRIDE   264                       // 256 bf16 + 8 pad
#define HBUF_BYTES (TILE_M * NF * 4)         // 262144: fp32 h_pre tile
#define AT_BYTES   (TILE_M * A_STRIDE * 2)   //   8448: bf16 A tile
#define TOPV_BYTES (TILE_M * TOPK * 4)       //   2048
#define TOPI_BYTES (TILE_M * TOPK * 4)       //   2048
#define SMEM_BYTES (HBUF_BYTES + AT_BYTES + TOPV_BYTES + TOPI_BYTES) // 274688 < 320KB

// ---------------- workspace layout ----------------
// Bpack: W_enc as bf16, pre-swizzled into per-lane WMMA B fragments.
//   fragment id = (ntile*8 + kt)*32 + lane ; each fragment = 16 bf16 = 32B.
//   lane n = lane&15, K = kt*32 + (lane>=16 ? 16 : 0) + 0..15
#define NTILES      (NF / 16)                            // 256
#define BPACK_ELEMS ((size_t)NTILES * 8 * 32 * 16)       // 1,048,576 bf16
#define BPACK_BYTES (BPACK_ELEMS * 2)                    // 2 MB
#define WDT_ELEMS   ((size_t)NF * DIN)                   // 1,048,576 f32
#define WDT_BYTES   (WDT_ELEMS * 4)                      // 4 MB
#define WS_NEEDED   (BPACK_BYTES + WDT_BYTES)            // 6 MB

// ============ prep kernel 1: W_enc (N,D) fp32 -> swizzled bf16 fragments ============
__global__ __launch_bounds__(256) void pack_wenc_kernel(
    const float* __restrict__ W_enc, __bf16* __restrict__ Bpack)
{
    const int frag = blockIdx.x * blockDim.x + threadIdx.x;  // [0, 65536)
    if (frag >= NTILES * 8 * 32) return;
    const int lane  = frag & 31;
    const int kt    = (frag >> 5) & 7;
    const int ntile = frag >> 8;
    const int n  = ntile * 16 + (lane & 15);
    const int kb = kt * 32 + (lane >> 4) * 16;
    const float* src = W_enc + (size_t)n * DIN + kb;         // 16 contiguous f32
    __bf16* dst = Bpack + (size_t)frag * 16;                 // 16 contiguous bf16
#pragma unroll
    for (int i = 0; i < 16; ++i) dst[i] = (__bf16)src[i];
}

// ============ prep kernel 2: W_dec (D,N) -> W_decT (N,D) fp32 ============
__global__ __launch_bounds__(256) void transpose_wdec_kernel(
    const float* __restrict__ W_dec, float* __restrict__ WdT)
{
    const int e = blockIdx.x * blockDim.x + threadIdx.x;     // [0, 1048576)
    const int n = e >> 8, d = e & 255;
    WdT[(size_t)n * DIN + d] = W_dec[(size_t)d * NF + n];
}

// ============ fused main kernel ============
template <bool PACKED>
__global__ __launch_bounds__(256) void sae_topk_fused(
    const float* __restrict__ x,     const float* __restrict__ W_enc,
    const float* __restrict__ b_enc, const float* __restrict__ W_dec,
    const float* __restrict__ b_dec, const float* __restrict__ mean,
    const float* __restrict__ stdv,
    const __bf16* __restrict__ Bpack, const float* __restrict__ WdT,
    float* __restrict__ xhat, float* __restrict__ hsp, float* __restrict__ hpre)
{
    extern __shared__ char smem[];
    float*  hbuf = (float*)smem;                                   // [16][4096]
    __bf16* aT   = (__bf16*)(smem + HBUF_BYTES);                   // [16][A_STRIDE]
    float*  topv = (float*)(smem + HBUF_BYTES + AT_BYTES);         // [16][32]
    int*    topi = (int*)(smem + HBUF_BYTES + AT_BYTES + TOPV_BYTES);

    const int tid  = threadIdx.x;
    const int lane = tid & 31;
    const int wave = tid >> 5;          // 8 waves
    const int row0 = blockIdx.x * TILE_M;

    // ---------- Phase 1: normalize + center, fp32 -> bf16 A tile ----------
    // x is streamed exactly once across the launch: non-temporal load keeps
    // the L2 free for the (reused) weight fragments.
    for (int e = tid; e < TILE_M * DIN; e += 256) {
        const int m = e >> 8, k = e & 255;
        const float xv = __builtin_nontemporal_load(x + (size_t)(row0 + m) * DIN + k);
        const float v  = (xv - mean[k]) / stdv[k] - b_dec[k];
        aT[m * A_STRIDE + k] = (__bf16)v;
    }
    __syncthreads();

    // ---------- Phase 2: h_pre = A * W_enc^T + b_enc via bf16 WMMA ----------
    // A layout (matches B fragments): lane m = lane&15; lanes 0-15 hold K 0..15,
    // lanes 16-31 hold K 16..31 (per 32-wide K step).
    const int mlane = lane & 15;
    const int hilo  = lane >> 4;
    const int kaddA = hilo * 16;

    for (int nt = 0; nt < NF / (8 * 16); ++nt) {     // 32 N-tiles per wave
        const int ncol = wave * (NF / 8) + nt * 16;  // this wave's tile base column
        const int gtile = ncol >> 4;                 // global 16-col tile index

        v8f acc = {};
#pragma unroll
        for (int kt = 0; kt < DIN / 32; ++kt) {      // K = 256 in 8 steps of 32
            const int k0 = kt * 32;
            // A fragment: two aligned 8x-bf16 chunks from LDS
            const __bf16* ap = aT + mlane * A_STRIDE + k0 + kaddA;
            const v8bf alo = *(const v8bf*)ap;
            const v8bf ahi = *(const v8bf*)(ap + 8);
            v16bf a;
#pragma unroll
            for (int i = 0; i < 8; ++i) { a[i] = alo[i]; a[i + 8] = ahi[i]; }

            v16bf b;
            if constexpr (PACKED) {
                // pre-swizzled bf16 fragment: one coalesced 32B load per lane
                b = *(const v16bf*)(Bpack + ((size_t)(gtile * 8 + kt) * 32 + lane) * 16);
            } else {
                // fallback: gather fp32 from W_enc and convert in-line
                const int kb = k0 + hilo * 16;
                const float4* wp = (const float4*)(W_enc + (size_t)(ncol + mlane) * DIN + kb);
                const float4 w0 = wp[0], w1 = wp[1], w2 = wp[2], w3 = wp[3];
                b[0]  = (__bf16)w0.x; b[1]  = (__bf16)w0.y; b[2]  = (__bf16)w0.z; b[3]  = (__bf16)w0.w;
                b[4]  = (__bf16)w1.x; b[5]  = (__bf16)w1.y; b[6]  = (__bf16)w1.z; b[7]  = (__bf16)w1.w;
                b[8]  = (__bf16)w2.x; b[9]  = (__bf16)w2.y; b[10] = (__bf16)w2.z; b[11] = (__bf16)w2.w;
                b[12] = (__bf16)w3.x; b[13] = (__bf16)w3.y; b[14] = (__bf16)w3.z; b[15] = (__bf16)w3.w;
            }

            acc = __builtin_amdgcn_wmma_f32_16x16x32_bf16(
                false, a, false, b, (short)0, acc, false, false);
        }
        // C/D layout: lanes 0-15 N=lane, M=r ; lanes 16-31 N=lane-16, M=8+r
        // Only write LDS here; coalesced global h_pre stream happens in phase 3.
        const int c  = ncol + mlane;
        const float be = b_enc[c];
#pragma unroll
        for (int r = 0; r < 8; ++r) {
            const int m = hilo * 8 + r;
            hbuf[m * NF + c] = acc[r] + be;
        }
    }
    __syncthreads();

    // ---------- Phase 3+4: per-row h_pre stream, top-k, h_sparse, decode ----------
    for (int rr = 0; rr < 2; ++rr) {
        const int r = wave + rr * 8;                 // 8 waves x 2 = 16 rows
        float* hrow = hbuf + r * NF;

        // stream h_pre row to HBM, fully coalesced (128B per instruction) and
        // non-temporal (write-once stream; don't evict L2-resident weights);
        // track the lane-local max on the fly
        float lmax = -__builtin_inff(); int larg = lane;
        for (int j = 0; j < NF / 32; ++j) {
            const int cc = j * 32 + lane;
            const float v = hrow[cc];
            __builtin_nontemporal_store(v, hpre + (size_t)(row0 + r) * NF + cc);
            if (v > lmax) { lmax = v; larg = cc; }
        }
        // 32 rounds of wave argmax; winner zaps + rescans its 128 values
        for (int it = 0; it < TOPK; ++it) {
            float bv = lmax; int bl = lane;
#pragma unroll
            for (int off = 16; off >= 1; off >>= 1) {
                const float ov = __shfl_xor(bv, off, 32);
                const int   ol = __shfl_xor(bl, off, 32);
                if (ov > bv || (ov == bv && ol < bl)) { bv = ov; bl = ol; }
            }
            if (lane == bl) {
                topv[r * TOPK + it] = lmax;
                topi[r * TOPK + it] = larg;
                hrow[larg] = -__builtin_inff();
                lmax = -__builtin_inff(); larg = lane;
                for (int j = 0; j < NF / 32; ++j) {
                    const int cc = j * 32 + lane;
                    const float v = hrow[cc];
                    if (v > lmax) { lmax = v; larg = cc; }
                }
            }
        }

        // h_sparse: zero row in LDS, scatter the 32 winners (same-wave LDS
        // ops are in-order, so zero-before-scatter holds), stream out coalesced
        for (int j = 0; j < NF / 32; ++j) hrow[j * 32 + lane] = 0.0f;
        hrow[topi[r * TOPK + lane]] = topv[r * TOPK + lane];
        for (int j = 0; j < NF / 32; ++j) {
            const int cc = j * 32 + lane;
            __builtin_nontemporal_store(hrow[cc], hsp + (size_t)(row0 + r) * NF + cc);
        }

        // decode: x_hat = sum_l v_l * W_dec[:, idx_l] + b_dec
        float acc[DIN / 32];
#pragma unroll
        for (int j = 0; j < DIN / 32; ++j) acc[j] = b_dec[lane + 32 * j];
        for (int l = 0; l < TOPK; ++l) {
            const float v  = topv[r * TOPK + l];
            const int  id  = topi[r * TOPK + l];
            if constexpr (PACKED) {
                const float* wr = WdT + (size_t)id * DIN;   // coalesced row
#pragma unroll
                for (int j = 0; j < DIN / 32; ++j)
                    acc[j] += v * wr[lane + 32 * j];
            } else {
                const float* wc = W_dec + id;               // strided column gather
#pragma unroll
                for (int j = 0; j < DIN / 32; ++j)
                    acc[j] += v * wc[(size_t)(lane + 32 * j) * NF];
            }
        }
#pragma unroll
        for (int j = 0; j < DIN / 32; ++j)
            __builtin_nontemporal_store(acc[j],
                xhat + (size_t)(row0 + r) * DIN + lane + 32 * j);
    }
}

extern "C" void kernel_launch(void* const* d_in, const int* in_sizes, int n_in,
                              void* d_out, int out_size, void* d_ws, size_t ws_size,
                              hipStream_t stream) {
    const float* x     = (const float*)d_in[0];
    const float* W_enc = (const float*)d_in[1];
    const float* b_enc = (const float*)d_in[2];
    const float* W_dec = (const float*)d_in[3];
    const float* b_dec = (const float*)d_in[4];
    const float* mean  = (const float*)d_in[5];
    const float* stdv  = (const float*)d_in[6];

    float* out  = (float*)d_out;
    float* xhat = out;                                  // (B, D)
    float* hsp  = out + (size_t)BATCH * DIN;            // (B, N)
    float* hpre = hsp + (size_t)BATCH * NF;             // (B, N)

    const dim3 grid(BATCH / TILE_M);   // 2048 workgroups
    const dim3 block(256);             // 8 wave32 per workgroup

    if (ws_size >= WS_NEEDED) {
        __bf16* Bpack = (__bf16*)d_ws;
        float*  WdT   = (float*)((char*)d_ws + BPACK_BYTES);
        // one-time weight repack (re-run each call for determinism; ~6MB traffic)
        pack_wenc_kernel<<<dim3(NTILES * 8 * 32 / 256), block, 0, stream>>>(W_enc, Bpack);
        transpose_wdec_kernel<<<dim3(WDT_ELEMS / 256), block, 0, stream>>>(W_dec, WdT);
        sae_topk_fused<true><<<grid, block, SMEM_BYTES, stream>>>(
            x, W_enc, b_enc, W_dec, b_dec, mean, stdv, Bpack, WdT, xhat, hsp, hpre);
    } else {
        sae_topk_fused<false><<<grid, block, SMEM_BYTES, stream>>>(
            x, W_enc, b_enc, W_dec, b_dec, mean, stdv, nullptr, nullptr, xhat, hsp, hpre);
    }
}
